// BitSwarmLinear_57492432224591
// MI455X (gfx1250) — compile-verified, matrix operand order checked by reading, not run on hardware
//
#include <hip/hip_runtime.h>
#include <hip/hip_bf16.h>
#include <cstdint>
#include <cstddef>

// BitSwarmLinear for MI455X (gfx1250, wave32, WMMA, WGP-shared LDS).
//   Stage 1: population (2048x2048x32 f32, 536 MB) -> sign-binarized bf16 +/-1
//            weights pre-swizzled into WMMA B-fragment tiles (8 MB in d_ws).
//   Stage 2: out = x @ W^T via v_wmma_f32_16x16x32_bf16 with double-buffered
//            LDS block tiling (128M x 256N per block); B tiles staged with the
//            gfx1250 async global->LDS path (ASYNCcnt tracked).

#define IN_F   2048
#define OUT_F  2048
#define MROWS  16384   // 4 * 4096
#define SWARM  32

typedef __attribute__((ext_vector_type(16))) __bf16 v16bf;
typedef __attribute__((ext_vector_type(8)))  float  v8f;
typedef __attribute__((ext_vector_type(4)))  int    v4i;

#if __has_builtin(__builtin_amdgcn_global_load_async_to_lds_b128)
#define HAVE_ASYNC_LDS 1
#else
#define HAVE_ASYNC_LDS 0
#endif

// b128 async copy operand types (from toolchain diagnostic: int4* in AS1/AS3)
#define TO_GLOBAL_B128(p) ((__attribute__((address_space(1))) v4i*)(p))
#define TO_LDS_B128(p)    ((__attribute__((address_space(3))) v4i*)(p))

static __device__ __forceinline__ void wait_asynccnt0() {
#if HAVE_ASYNC_LDS
#if __has_builtin(__builtin_amdgcn_s_wait_asynccnt)
    __builtin_amdgcn_s_wait_asynccnt(0);
#else
    asm volatile("s_wait_asynccnt 0x0" ::: "memory");
#endif
#endif
}

static __device__ __forceinline__ uint32_t pack_bf16x2(float a, float b) {
    // lowers to v_cvt_pk_bf16_f32
    uint16_t ua = __builtin_bit_cast(uint16_t, (__bf16)a);
    uint16_t ub = __builtin_bit_cast(uint16_t, (__bf16)b);
    return (uint32_t)ua | ((uint32_t)ub << 16);
}

// ---------------------------------------------------------------------------
// Fragment tile layout for V_WMMA_F32_16X16X32_BF16 (ISA 7.12.2, 16-bit).
// One 16x32 tile = 32 lanes x 16 bf16, 32 contiguous bytes per lane:
//   B tile (N=16 x K=32):
//     lanes 0-15  (col n=l)    : j=0..7 -> K 0..7,  j=8..15 -> K 16..23
//     lanes 16-31 (col n=l-16) : j=0..7 -> K 8..15, j=8..15 -> K 24..31
//   A tile (M=16 x K=32): identical pattern with rows in place of cols.
// ---------------------------------------------------------------------------

__global__ void bitswarm_pack_w(const float* __restrict__ pop,
                                uint32_t* __restrict__ wtiles) {
    // One thread per (o, even-k pair): read 64 contiguous floats (256 B,
    // perfectly streaming), write one packed dword into the fragment tile.
    uint32_t tid = blockIdx.x * blockDim.x + threadIdx.x;
    uint32_t o  = tid / (IN_F / 2);
    uint32_t kk = (tid % (IN_F / 2)) * 2u;

    const float4* p = (const float4*)(pop + ((size_t)o * IN_F + kk) * SWARM);
    float s0 = 0.f, s1 = 0.f;
#pragma unroll
    for (int i = 0; i < 8; ++i) { float4 v = p[i];     s0 += (v.x + v.y) + (v.z + v.w); }
#pragma unroll
    for (int i = 0; i < 8; ++i) { float4 v = p[8 + i]; s1 += (v.x + v.y) + (v.z + v.w); }

    // sign with sign(0) -> +1 ; bf16(+1)=0x3F80, bf16(-1)=0xBF80
    uint32_t b0 = (s0 < 0.f) ? 0xBF80u : 0x3F80u;
    uint32_t b1 = (s1 < 0.f) ? 0xBF80u : 0x3F80u;

    uint32_t to  = o >> 4,  n   = o & 15u;
    uint32_t tk  = kk >> 5, kin = kk & 31u;
    uint32_t hi  = kin >> 3;
    uint32_t lane = n + 16u * (hi & 1u);
    uint32_t j    = (kin & 7u) + 8u * (hi >> 1);
    size_t   tile = (size_t)to * (IN_F / 32) + tk;
    // tile = 512 bf16 = 256 dwords ; lane slice = 8 dwords ; j is even.
    wtiles[tile * 256 + lane * 8 + (j >> 1)] = b0 | (b1 << 16);
}

// ---------------------------------------------------------------------------
// GEMM. Block tile: 128(M) x 256(N), 8 waves as 2(M) x 4(N); each wave owns a
// 64x64 register tile = 4x4 WMMA tiles. K loop in chunks of 32 with
// double-buffered LDS: A tile 8 KB (8 frag-tiles) + B tile 16 KB (16 tiles).
// Grid = (16384/128, 2048/256) = (128, 8).
// ---------------------------------------------------------------------------

#define KC      32
#define NKC     (IN_F / KC)        // 64 K-chunks
#define A_TILES 8                  // 128 M rows  -> 8 fragment tiles
#define B_TILES 16                 // 256 N cols  -> 16 fragment tiles

__global__ void __launch_bounds__(256, 1) bitswarm_gemm(
    const float* __restrict__ x,
    const uint32_t* __restrict__ wtiles,
    float* __restrict__ out) {

    __shared__ uint32_t ldsA[2][A_TILES * 256];   // 2 x 8 KB
    __shared__ uint32_t ldsB[2][B_TILES * 256];   // 2 x 16 KB

    const int tid  = threadIdx.x;
    const int lane = tid & 31;
    const int wave = tid >> 5;
    const int wm = wave >> 2;                      // 0..1 (M)
    const int wn = wave & 3;                       // 0..3 (N)
    const int blk_m = blockIdx.x * 128;
    const int blk_n = blockIdx.y * 256;

    // --- staging thread roles -------------------------------------------------
    // A: thread (mt = tid>>5, l = tid&31) builds one lane-slice (32 B) of one
    //    A fragment tile: row = 16*mt + (l&15), K runs {b1..b1+7, b1+16..b1+23},
    //    b1 = (l<16) ? 0 : 8.
    const int a_mt   = tid >> 5;
    const int a_row  = lane & 15;
    const int a_koff = (lane >> 4) ? 8 : 0;
    const float* a_src = x + (size_t)(blk_m + a_mt * 16 + a_row) * IN_F + a_koff;
    uint32_t* const a_dst0 = &ldsA[0][a_mt * 256 + lane * 8];
    uint32_t* const a_dst1 = &ldsA[1][a_mt * 256 + lane * 8];

    // B: already fragment-packed in global; raw copy. 512 lane-slices, 2/thread.
    const int b_t0 = tid >> 5;          // tiles 0..7
    const int b_t1 = b_t0 + 8;          // tiles 8..15

    // stage K-chunk kc into buffer buf
    auto stage = [&](int kc, int buf) {
        const int k = kc * KC;
        // ---- A: fp32 -> bf16 fragment pack (needs VALU convert; manual path) ----
        const float4* xp = (const float4*)(a_src + k);
        float4 q0 = xp[0], q1 = xp[1], q2 = xp[4], q3 = xp[5];
        uint32_t d0 = pack_bf16x2(q0.x, q0.y);
        uint32_t d1 = pack_bf16x2(q0.z, q0.w);
        uint32_t d2 = pack_bf16x2(q1.x, q1.y);
        uint32_t d3 = pack_bf16x2(q1.z, q1.w);
        uint32_t d4 = pack_bf16x2(q2.x, q2.y);
        uint32_t d5 = pack_bf16x2(q2.z, q2.w);
        uint32_t d6 = pack_bf16x2(q3.x, q3.y);
        uint32_t d7 = pack_bf16x2(q3.z, q3.w);
        uint4* ad = (uint4*)(buf ? a_dst1 : a_dst0);
        ad[0] = make_uint4(d0, d1, d2, d3);
        ad[1] = make_uint4(d4, d5, d6, d7);
        // ---- B: verbatim copy of pre-swizzled fragment slices ----
        const size_t tk = (size_t)(k >> 5);
        const size_t tb = (size_t)(blk_n >> 4);
        const uint4* s0 = (const uint4*)(wtiles + ((tb + b_t0) * (IN_F / 32) + tk) * 256 + lane * 8);
        const uint4* s1 = (const uint4*)(wtiles + ((tb + b_t1) * (IN_F / 32) + tk) * 256 + lane * 8);
        uint4* bd0 = (uint4*)&ldsB[buf][b_t0 * 256 + lane * 8];
        uint4* bd1 = (uint4*)&ldsB[buf][b_t1 * 256 + lane * 8];
#if HAVE_ASYNC_LDS
        // gfx1250 async global->LDS: no VGPR round-trip, tracked by ASYNCcnt.
        // imm offset applies to both global and LDS addresses (ISA 10.7.4).
        __builtin_amdgcn_global_load_async_to_lds_b128(TO_GLOBAL_B128(s0), TO_LDS_B128(bd0), 0, 0);
        __builtin_amdgcn_global_load_async_to_lds_b128(TO_GLOBAL_B128(s0), TO_LDS_B128(bd0), 16, 0);
        __builtin_amdgcn_global_load_async_to_lds_b128(TO_GLOBAL_B128(s1), TO_LDS_B128(bd1), 0, 0);
        __builtin_amdgcn_global_load_async_to_lds_b128(TO_GLOBAL_B128(s1), TO_LDS_B128(bd1), 16, 0);
#else
        bd0[0] = s0[0]; bd0[1] = s0[1];
        bd1[0] = s1[0]; bd1[1] = s1[1];
#endif
    };

    v8f acc[4][4];
#pragma unroll
    for (int i = 0; i < 4; ++i)
#pragma unroll
        for (int t = 0; t < 4; ++t)
            acc[i][t] = (v8f){0.f, 0.f, 0.f, 0.f, 0.f, 0.f, 0.f, 0.f};

    stage(0, 0);
    wait_asynccnt0();
    __syncthreads();

    for (int kc = 0; kc < NKC; ++kc) {
        const int cur = kc & 1;
        if (kc + 1 < NKC) stage(kc + 1, cur ^ 1);   // overlap next-chunk staging

        // fragment fetch: one contiguous 32 B LDS read per lane per tile
        v16bf a[4], b[4];
#pragma unroll
        for (int i = 0; i < 4; ++i)
            a[i] = *((const v16bf*)&ldsA[cur][(wm * 4 + i) * 256] + lane);
#pragma unroll
        for (int t = 0; t < 4; ++t)
            b[t] = *((const v16bf*)&ldsB[cur][(wn * 4 + t) * 256] + lane);

#pragma unroll
        for (int i = 0; i < 4; ++i)
#pragma unroll
            for (int t = 0; t < 4; ++t)
                acc[i][t] = __builtin_amdgcn_wmma_f32_16x16x32_bf16(
                    /*neg_a=*/false, a[i], /*neg_b=*/false, b[t],
                    /*c_mod=*/(short)0, acc[i][t],
                    /*reuse_a=*/false, /*reuse_b=*/false);

        wait_asynccnt0();  // async copies into (cur^1) done before publishing
        __syncthreads();   // all reads of `cur` done before it is restaged
    }

    // C/D layout: VGPR v, lanes 0-15 -> (row v, col lane), lanes 16-31 -> (row v+8, col lane-16)
    const int crow = (lane >> 4) * 8;
    const int ccol = lane & 15;
#pragma unroll
    for (int i = 0; i < 4; ++i)
#pragma unroll
        for (int t = 0; t < 4; ++t) {
            float* op = out + (size_t)(blk_m + wm * 64 + i * 16 + crow) * OUT_F
                            + (blk_n + wn * 64 + t * 16 + ccol);
#pragma unroll
            for (int v = 0; v < 8; ++v)
                op[(size_t)v * OUT_F] = acc[i][t][v];
        }
}

extern "C" void kernel_launch(void* const* d_in, const int* in_sizes, int n_in,
                              void* d_out, int out_size, void* d_ws, size_t ws_size,
                              hipStream_t stream) {
    const float* x   = (const float*)d_in[0];   // (4, 4096, 2048) f32
    const float* pop = (const float*)d_in[1];   // (2048, 2048, 32) f32
    float* out = (float*)d_out;                 // (4, 4096, 2048) f32
    uint32_t* wtiles = (uint32_t*)d_ws;         // 8 MB of packed bf16 B-fragments

    // Stage 1: 2048*2048/2 threads, each handles one (o, k-pair).
    {
        int threads = 256;
        int total = (OUT_F * IN_F) / 2;
        bitswarm_pack_w<<<total / threads, threads, 0, stream>>>(pop, wtiles);
    }
    // Stage 2: LDS-tiled WMMA GEMM (async B staging when available).
    {
        dim3 grid(MROWS / 128, OUT_F / 256);
        bitswarm_gemm<<<grid, 256, 0, stream>>>(x, wtiles, out);
    }
}
